// SelfAttentionLayerSingleMove_18657337933944
// MI455X (gfx1250) — matrix operand order repaired and checked by reading, not compiled, not run on hardware
//
#include <hip/hip_runtime.h>

typedef __attribute__((ext_vector_type(16))) _Float16 v16h;
typedef __attribute__((ext_vector_type(8)))  float    v8f;
typedef __attribute__((ext_vector_type(4)))  float    v4f;

#define DIM   256          // in/cmp/out dims all 256
#define MROWS 2048         // B * N = 8 * 256 flattened rows

// ---------------------------------------------------------------------------
// Stage 1: Out[M=2048, 256] = X[2048,256] @ W[256,256]^T + bias, via
// v_wmma_f32_16x16x32_f16. One wave per 16x16 output tile, 8 K-steps of 32.
// Tiles: 128 (M) x 16 (N) = 2048 tiles; 8 waves/block -> 256 blocks.
// ---------------------------------------------------------------------------
__global__ __launch_bounds__(256) void proj_gemm_f16wmma(
    const float* __restrict__ X,
    const float* __restrict__ W,
    const float* __restrict__ bias,
    float* __restrict__ Out)
{
    const int lane = threadIdx.x & 31;
    const int wave = threadIdx.x >> 5;
    const int tile = blockIdx.x * 8 + wave;   // 0..2047
    const int tm   = tile >> 4;               // 0..127
    const int tn   = tile & 15;               // 0..15
    const int half = lane >> 4;               // 0 | 1
    const int l16  = lane & 15;

    // A: row m of X; B: col n of W^T == row n of W (both row-major, len 256)
    const float* rowA = X + (tm * 16 + l16) * DIM;
    const float* rowB = W + (tn * 16 + l16) * DIM;
    const int offA = half * 8;    // A halves: K in {off..off+7} U {off+16..off+23}
    const int offB = half * 8;    // B halves: K in {off..off+7} U {off+16..off+23}

    v8f acc = {};
    #pragma unroll
    for (int ks = 0; ks < 8; ++ks) {
        const int kb = ks * 32;
        // A-operand fragments (4x b128 loads)
        v4f a0 = *(const v4f*)(rowA + kb + offA);
        v4f a1 = *(const v4f*)(rowA + kb + offA + 4);
        v4f a2 = *(const v4f*)(rowA + kb + offA + 16);
        v4f a3 = *(const v4f*)(rowA + kb + offA + 20);
        // B-operand fragments (4x b128 loads, same K split as A)
        v4f b0 = *(const v4f*)(rowB + kb + offB);
        v4f b1 = *(const v4f*)(rowB + kb + offB + 4);
        v4f b2 = *(const v4f*)(rowB + kb + offB + 16);
        v4f b3 = *(const v4f*)(rowB + kb + offB + 20);

        v16h A, Bm;
        #pragma unroll
        for (int i = 0; i < 4; ++i) {
            A[i]       = (_Float16)a0[i];
            A[4 + i]   = (_Float16)a1[i];
            A[8 + i]   = (_Float16)a2[i];
            A[12 + i]  = (_Float16)a3[i];
            Bm[i]      = (_Float16)b0[i];
            Bm[4 + i]  = (_Float16)b1[i];
            Bm[8 + i]  = (_Float16)b2[i];
            Bm[12 + i] = (_Float16)b3[i];
        }
        // D = A x B + C, fp32 accumulate
        acc = __builtin_amdgcn_wmma_f32_16x16x32_f16(
                  false, A, false, Bm, (short)0, acc, false, false);
    }

    const int   col = tn * 16 + l16;
    const float bb  = bias[col];
    const int   r0  = tm * 16 + half * 8;     // C/D: VGPR j -> row j + 8*half
    #pragma unroll
    for (int j = 0; j < 8; ++j)
        Out[(r0 + j) * DIM + col] = acc[j] + bb;
}

// ---------------------------------------------------------------------------
// Stage 2: per-cell neighborhood attention with online softmax.
// One wave32 per (b,n) cell; lane owns channels [lane*8, lane*8+8).
// scores = (q . k_nbr)/16, masked; out = softmax(scores) @ v_nbr.
// ---------------------------------------------------------------------------
__global__ __launch_bounds__(256) void attn_online_softmax(
    const float* __restrict__ qb,
    const float* __restrict__ kb,
    const float* __restrict__ vb,
    const int* __restrict__ nbrs,
    const unsigned char* __restrict__ mask,   // jax bool -> 1 byte
    float* __restrict__ out,
    int K)
{
    const int lane = threadIdx.x & 31;
    const int wave = threadIdx.x >> 5;
    const int cell = blockIdx.x * 8 + wave;   // 0..2047 == b*256 + n
    const int b    = cell >> 8;
    const int n    = cell & 255;

    const float* qrow = qb + cell * DIM + lane * 8;
    const v4f q0 = *(const v4f*)qrow;
    const v4f q1 = *(const v4f*)(qrow + 4);

    float m = -3.0e38f;
    float l = 0.0f;
    v4f acc0 = {};
    v4f acc1 = {};

    for (int kk = 0; kk < K; ++kk) {
        // wave-uniform predicate: skip padded neighbor slots entirely
        if (!mask[n * K + kk]) continue;
        const int nb = nbrs[n * K + kk];

        const float* krow = kb + (b * 256 + nb) * DIM + lane * 8;
        v4f k0 = *(const v4f*)krow;
        v4f k1 = *(const v4f*)(krow + 4);

        float s = q0[0]*k0[0] + q0[1]*k0[1] + q0[2]*k0[2] + q0[3]*k0[3]
                + q1[0]*k1[0] + q1[1]*k1[1] + q1[2]*k1[2] + q1[3]*k1[3];
        // wave32 butterfly reduce -> every lane holds the full dot
        #pragma unroll
        for (int off = 16; off > 0; off >>= 1)
            s += __shfl_xor(s, off, 32);
        s *= 0.0625f;   // 1/sqrt(256)

        const float mn = fmaxf(m, s);
        const float sc = __expf(m - mn);   // rescale old accumulator
        const float p  = __expf(s - mn);   // weight of this neighbor

        const float* vrow = vb + (b * 256 + nb) * DIM + lane * 8;
        v4f v0 = *(const v4f*)vrow;
        v4f v1 = *(const v4f*)(vrow + 4);

        l = l * sc + p;
        #pragma unroll
        for (int i = 0; i < 4; ++i) {
            acc0[i] = acc0[i] * sc + p * v0[i];
            acc1[i] = acc1[i] * sc + p * v1[i];
        }
        m = mn;
    }

    const float inv = 1.0f / l;   // every cell has >= 45 valid neighbors
    float* orow = out + cell * DIM + lane * 8;
    *(v4f*)orow       = acc0 * inv;
    *(v4f*)(orow + 4) = acc1 * inv;
}

// ---------------------------------------------------------------------------
// Inputs (setup_inputs order): query_X, key_X, value_X, Wq, bq, Wk, bk,
//                              Wv, bv, neighbors, mask
// ---------------------------------------------------------------------------
extern "C" void kernel_launch(void* const* d_in, const int* in_sizes, int n_in,
                              void* d_out, int out_size, void* d_ws, size_t ws_size,
                              hipStream_t stream) {
    const float* qX = (const float*)d_in[0];
    const float* kX = (const float*)d_in[1];
    const float* vX = (const float*)d_in[2];
    const float* Wq = (const float*)d_in[3];
    const float* bq = (const float*)d_in[4];
    const float* Wk = (const float*)d_in[5];
    const float* bk = (const float*)d_in[6];
    const float* Wv = (const float*)d_in[7];
    const float* bv = (const float*)d_in[8];
    const int*   nb = (const int*)d_in[9];
    const unsigned char* mk = (const unsigned char*)d_in[10];
    float* out = (float*)d_out;

    const int K = in_sizes[9] / 256;   // neighbors is [256, Kmax] (Kmax = 95)

    // workspace: q | k | v, each 2048x256 fp32 (6 MB total)
    float* q = (float*)d_ws;
    float* k = q + MROWS * DIM;
    float* v = k + MROWS * DIM;

    proj_gemm_f16wmma<<<256, 256, 0, stream>>>(qX, Wq, bq, q);
    proj_gemm_f16wmma<<<256, 256, 0, stream>>>(kX, Wk, bk, k);
    proj_gemm_f16wmma<<<256, 256, 0, stream>>>(vX, Wv, bv, v);
    attn_online_softmax<<<256, 256, 0, stream>>>(q, k, v, nb, mk, out, K);
}
